// Encoder_28595892257001
// MI455X (gfx1250) — matrix operand order repaired and checked by reading, not compiled, not run on hardware
//
#include <hip/hip_runtime.h>
#include <hip/hip_bf16.h>
#include <stdint.h>

// CDNA5 / gfx1250 tree-encoder, mixed precision: bf16 activations/weights,
// V_WMMA_F32_16X16X32_BF16 with f32 accumulation, f32 bias/PReLU epilogue.
// Staging uses GLOBAL_LOAD_ASYNC_TO_LDS_B128 (ASYNCcnt) with double-buffered
// LDS so global->LDS traffic overlaps the WMMA stream.

typedef __bf16 bf16_t;
typedef __bf16 v16bf __attribute__((ext_vector_type(16)));
typedef float  v8f   __attribute__((ext_vector_type(8)));

#define ENC_DIM   256
#define ENC_KTOT  512
#define ENC_KC    64     // K-chunk staged in LDS (2 WMMA K-steps)
#define ENC_NCHUNK (ENC_KTOT / ENC_KC)
#define ENC_MBLK  32
#define AS_S      72     // 64 + pad (bf16 units): 16B-aligned b128 frags, bank spread
#define BS_S      72

// ---- CDNA5 async global->LDS copy (tracked by ASYNCcnt) -------------------
__device__ __forceinline__ void async_copy_b128(unsigned lds_off, const void* gptr) {
    asm volatile("global_load_async_to_lds_b128 %0, %1, off"
                 :: "v"(lds_off), "v"(gptr) : "memory");
}
__device__ __forceinline__ void wait_async0() {
    asm volatile("s_wait_asynccnt 0" ::: "memory");
}
__device__ __forceinline__ unsigned lds_off32(const void* p) {
    // Flat LDS addresses carry the LDS offset in the low 32 bits (ISA §10.2).
    return (unsigned)(size_t)p;
}

// ---------------------------------------------------------------------------
// threefry2x32 (JAX PRNG) to reproduce _child_s indices on-device.
// ---------------------------------------------------------------------------
__device__ __forceinline__ unsigned rotl32(unsigned v, int r) {
    return (v << r) | (v >> (32 - r));
}

__device__ __forceinline__ void threefry2x32(unsigned k0, unsigned k1,
                                             unsigned x0, unsigned x1,
                                             unsigned& o0, unsigned& o1) {
    unsigned ks2 = k0 ^ k1 ^ 0x1BD11BDAu;
    x0 += k0; x1 += k1;
#define TFR(r) { x0 += x1; x1 = rotl32(x1, (r)); x1 ^= x0; }
    TFR(13) TFR(15) TFR(26) TFR(6)
    x0 += k1;  x1 += ks2 + 1u;
    TFR(17) TFR(29) TFR(16) TFR(24)
    x0 += ks2; x1 += k0 + 2u;
    TFR(13) TFR(15) TFR(26) TFR(6)
    x0 += k0;  x1 += k1 + 3u;
    TFR(17) TFR(29) TFR(16) TFR(24)
    x0 += k1;  x1 += ks2 + 4u;
#undef TFR
    o0 = x0; o1 = x1;
}

// cs = randint(fold_in(key(42), i), (n,), 0, 8n); power-of-two span -> mask.
__global__ void cs_kernel(int* __restrict__ cs, unsigned lvl_i, int n, unsigned mask) {
    int r = blockIdx.x * blockDim.x + threadIdx.x;
    if (r >= n) return;
    unsigned f0, f1;
    threefry2x32(0u, 42u, 0u, lvl_i, f0, f1);   // fold_in(key(42), i)
    unsigned o0, o1, bits;
    if (n == 1) {                               // odd size: zero-padded pair
        threefry2x32(f0, f1, 0u, 0u, o0, o1);
        bits = o0;
    } else {
        int half = n >> 1;
        int p = (r < half) ? r : r - half;
        threefry2x32(f0, f1, (unsigned)p, (unsigned)(p + half), o0, o1);
        bits = (r < half) ? o0 : o1;
    }
    cs[r] = (int)(bits & mask);
}

// ---------------------------------------------------------------------------
// One-time: W[lvl][k][n] f32  ->  Wt[lvl][n][k] bf16  (transposed, n-major so
// B-fragments become contiguous 16B K-strips).
// ---------------------------------------------------------------------------
__global__ __launch_bounds__(256)
void convert_weights(const float* __restrict__ W, bf16_t* __restrict__ Wt, int total) {
    int idx = blockIdx.x * blockDim.x + threadIdx.x;
    if (idx >= total) return;
    const int per = ENC_KTOT * ENC_DIM;
    int l   = idx / per;
    int rem = idx - l * per;
    int k   = rem / ENC_DIM;
    int n   = rem - k * ENC_DIM;
    Wt[(size_t)l * per + (size_t)n * ENC_KTOT + k] = (bf16_t)W[idx];
}

// ---------------------------------------------------------------------------
// Leaf: out[row][c] = prelu(x[row]·W_leaf[:,c] + b_leaf[c]) in f32, store bf16.
// ---------------------------------------------------------------------------
__global__ __launch_bounds__(ENC_DIM)
void leaf_kernel(const float* __restrict__ x, const float* __restrict__ Wl,
                 const float* __restrict__ bl, const float* __restrict__ al,
                 bf16_t* __restrict__ out) {
    const int row = blockIdx.x;
    const int col = threadIdx.x;
    const float x0 = x[(size_t)row * 3 + 0];
    const float x1 = x[(size_t)row * 3 + 1];
    const float x2 = x[(size_t)row * 3 + 2];
    float v = fmaf(x0, Wl[col], fmaf(x1, Wl[ENC_DIM + col],
              fmaf(x2, Wl[2 * ENC_DIM + col], bl[col])));
    const float a = al[0];
    v = (v >= 0.0f) ? v : a * v;
    out[(size_t)row * ENC_DIM + col] = (bf16_t)v;
}

// ---------------------------------------------------------------------------
// Fragment load helper: two contiguous 16B LDS reads -> one v16bf operand.
// ---------------------------------------------------------------------------
__device__ __forceinline__ v16bf load_frag(const bf16_t* p, int second_off) {
    union { v16bf v; float4 f[2]; } u;
    u.f[0] = *(const float4*)p;
    u.f[1] = *(const float4*)(p + second_off);
    return u.v;
}

// ---------------------------------------------------------------------------
// Level GEMM: out(M x 256) = prelu(Acat(M x 512) @ W + bias), bf16 in, f32 acc.
// Acat K-halves:
//   plain (cs==nullptr): row r -> A + r*512 (contiguous reinterpret of
//                        (B, 2n, 256) as (B*n, 512))
//   sampled:             half0 = h + r*256;  half1 = S[(b*8n + cs[node])*256]
// Wt is 256x512 bf16 (n-major), so B-frag K-strips are contiguous.
// ---------------------------------------------------------------------------
template <bool OUT_F32>
__global__ __launch_bounds__(256)
void tree_gemm(const bf16_t* __restrict__ A,    // plain: prev level; sampled: h
               const bf16_t* __restrict__ S,    // sampled: 2-levels-up backup
               const int*    __restrict__ cs,   // nullptr => plain level
               int n_nodes, int samp_nodes,
               const bf16_t* __restrict__ Wt,   // 256 x 512 bf16
               const float*  __restrict__ bias, // 256 f32
               const float*  __restrict__ alpha_p,
               void* __restrict__ out_,         // bf16 level buffer, or f32 d_out
               int M) {
    __shared__ bf16_t As[2][ENC_MBLK * AS_S];
    __shared__ bf16_t Bs[2][ENC_DIM * BS_S];

    const int tid  = threadIdx.x;
    const int lane = tid & 31;
    const int wave = tid >> 5;
    const int wm   = wave >> 2;      // wave M-tile: 0..1
    const int wn   = wave & 3;       // wave N-tile: 0..3
    const int lm   = lane & 15;
    const int hi   = lane >> 4;      // half-wave: selects K sub-strips / D rows
    const int m0   = blockIdx.x * ENC_MBLK;

    v8f acc[4] = {};

    // Cooperative A-load coords: 32 rows x 64 bf16 per chunk, 8 bf16/thread.
    const int arow = tid >> 3;             // 0..31
    const int acol = (tid & 7) * 8;        // 16B group
    int r = m0 + arow; if (r > M - 1) r = M - 1;    // clamp: EXEC stays full

    const bf16_t* h0base;
    const bf16_t* h1base;
    if (cs == nullptr) {
        h0base = A + (size_t)r * (2 * ENC_DIM);
        h1base = h0base + ENC_DIM;
    } else {
        h0base = A + (size_t)r * ENC_DIM;
        const int b    = r / n_nodes;
        const int node = r - b * n_nodes;
        h1base = S + ((size_t)b * samp_nodes + cs[node]) * ENC_DIM;
    }

    // Async-stage one K-chunk into LDS buffer `buf` (no wait here).
    auto stage = [&](int kc, int buf) {
        {   // A tile: chunk never straddles the 256 half-boundary (64 | 256).
            const int kk = kc + acol;
            const bf16_t* src = (kk < ENC_DIM) ? (h0base + kk)
                                               : (h1base + (kk - ENC_DIM));
            async_copy_b128(lds_off32(&As[buf][arow * AS_S + acol]), src);
        }
#pragma unroll
        for (int rep = 0; rep < 8; ++rep) {   // Wt tile: 256 x 64 bf16
            const int row = rep * 32 + (tid >> 3);
            const int col = (tid & 7) * 8;
            async_copy_b128(lds_off32(&Bs[buf][row * BS_S + col]),
                            &Wt[(size_t)row * ENC_KTOT + kc + col]);
        }
    };

    stage(0, 0);
    wait_async0();
    __syncthreads();

#pragma unroll
    for (int c = 0; c < ENC_NCHUNK; ++c) {
        const int cur = c & 1;
        if (c + 1 < ENC_NCHUNK) stage((c + 1) * ENC_KC, cur ^ 1);  // prefetch

#pragma unroll
        for (int k0 = 0; k0 < ENC_KC; k0 += 32) {
            // A frag 16x32 bf16: VGPR0-3 <- K=[k0+8*hi,+8), VGPR4-7 <- +16.
            const v16bf af =
                load_frag(&As[cur][(wm * 16 + lm) * AS_S + k0 + 8 * hi], 16);
            // Preload all 4 B frags (same K order as A frag), then batch WMMAs.
            v16bf bf[4];
#pragma unroll
            for (int t = 0; t < 4; ++t)
                bf[t] = load_frag(
                    &Bs[cur][(wn * 64 + t * 16 + lm) * BS_S + k0 + 8 * hi], 16);
#pragma unroll
            for (int t = 0; t < 4; ++t)
                acc[t] = __builtin_amdgcn_wmma_f32_16x16x32_bf16(
                    false, af, false, bf[t], (short)0, acc[t], false, false);
        }

        wait_async0();      // my prefetch for chunk c+1 has landed in LDS
        __syncthreads();    // everyone's prefetch landed; safe to flip buffers
    }

    // Epilogue: f32 bias + PReLU; D layout: row = g + 8*hi, col = lm.
    const float alpha = *alpha_p;
    const int orow0 = m0 + wm * 16 + 8 * hi;
#pragma unroll
    for (int t = 0; t < 4; ++t) {
        const int col = wn * 64 + t * 16 + lm;
        const float bb = bias[col];
#pragma unroll
        for (int g = 0; g < 8; ++g) {
            const int row = orow0 + g;
            float v = acc[t][g] + bb;
            v = (v >= 0.0f) ? v : alpha * v;
            if (row < M) {
                if (OUT_F32)
                    ((float*)out_)[(size_t)row * ENC_DIM + col] = v;
                else
                    ((bf16_t*)out_)[(size_t)row * ENC_DIM + col] = (bf16_t)v;
            }
        }
    }
}

// ---------------------------------------------------------------------------
// Driver.  Workspace (bytes): 13 bf16 level buffers (~134 MB, L2-friendly) +
// bf16 h scratch + transposed bf16 weights + cs ints.
// ---------------------------------------------------------------------------
extern "C" void kernel_launch(void* const* d_in, const int* in_sizes, int n_in,
                              void* d_out, int out_size, void* d_ws, size_t ws_size,
                              hipStream_t stream) {
    const float* x      = (const float*)d_in[0];
    const float* W_leaf = (const float*)d_in[1];
    const float* b_leaf = (const float*)d_in[2];
    const float* a_leaf = (const float*)d_in[3];
    const float* Wc     = (const float*)d_in[4];
    const float* bc     = (const float*)d_in[5];
    const float* ac     = (const float*)d_in[6];
    const float* Ws     = (const float*)d_in[7];
    const float* bs     = (const float*)d_in[8];

    const int B = 16, L = 13, SL = 3;
    const size_t WPER = (size_t)ENC_KTOT * ENC_DIM;   // elems per weight matrix

    char* ws = (char*)d_ws;
    size_t off = 0;
    bf16_t* lvl[13];
    for (int k = 0; k < L; ++k) {
        lvl[k] = (bf16_t*)(ws + off);
        off += (size_t)B * ((size_t)1 << (L - k)) * ENC_DIM * sizeof(bf16_t);
    }
    bf16_t* htmp = (bf16_t*)(ws + off);
    off += (size_t)B * 1024 * ENC_DIM * sizeof(bf16_t);
    bf16_t* WcT = (bf16_t*)(ws + off); off += 13 * WPER * sizeof(bf16_t);
    bf16_t* WsT = (bf16_t*)(ws + off); off += 11 * WPER * sizeof(bf16_t);
    int* csbuf = (int*)(ws + off);

    // One-time weight convert+transpose (f32 -> bf16, n-major).
    {
        int t1 = (int)(13 * WPER);
        convert_weights<<<(t1 + 255) / 256, 256, 0, stream>>>(Wc, WcT, t1);
        int t2 = (int)(11 * WPER);
        convert_weights<<<(t2 + 255) / 256, 256, 0, stream>>>(Ws, WsT, t2);
    }

    // Leaf layer -> lvl[0] (bf16).
    const int leaf_rows = B << L;      // 131072
    leaf_kernel<<<leaf_rows, ENC_DIM, 0, stream>>>(x, W_leaf, b_leaf, a_leaf, lvl[0]);

    for (int j = 0; j < L; ++j) {
        const int i = L - 1 - j;
        const int n = 1 << i;
        const int M = B * n;
        const bool samp = (i <= L - SL);
        const int grid = (M + ENC_MBLK - 1) / ENC_MBLK;

        bf16_t* h_out = samp ? htmp : lvl[j + 1];
        tree_gemm<false><<<grid, 256, 0, stream>>>(
            lvl[j], lvl[j], nullptr, n, 0,
            WcT + (size_t)j * WPER, bc + (size_t)j * ENC_DIM,
            ac + j, (void*)h_out, M);

        if (samp) {
            cs_kernel<<<(n + 255) / 256, 256, 0, stream>>>(
                csbuf, (unsigned)i, n, (unsigned)(8 * n - 1));
            const int js = j - (SL - 1);
            const bf16_t* WT = WsT + (size_t)js * WPER;
            const float*  bb = bs + (size_t)js * ENC_DIM;
            if (j == L - 1) {
                tree_gemm<true><<<grid, 256, 0, stream>>>(
                    htmp, lvl[j - 2], csbuf, n, 8 * n, WT, bb, ac + j,
                    d_out, M);
            } else {
                tree_gemm<false><<<grid, 256, 0, stream>>>(
                    htmp, lvl[j - 2], csbuf, n, 8 * n, WT, bb, ac + j,
                    (void*)lvl[j + 1], M);
            }
        }
    }
}
